// SSMFusionModule_58643483459626
// MI455X (gfx1250) — matrix operand order repaired and checked by reading, not compiled, not run on hardware
//
#include <hip/hip_runtime.h>
#include <hip/hip_bf16.h>
#include <math.h>

// ---------------------------------------------------------------------------
// Problem dims
// ---------------------------------------------------------------------------
#define B_       4
#define L_       1024
#define M_TOT    (B_ * L_)          // 4096 token rows
#define D_MODEL_ 768
#define D_INNER_ 1536
#define D_STATE_ 16
#define D_CONV_  4
#define DT_RANK_ 48
#define XDBL_N   (DT_RANK_ + 2 * D_STATE_)   // 80
#define XZ_N     (2 * D_INNER_)              // 3072

typedef __attribute__((ext_vector_type(16))) _Float16 v16h;
typedef __attribute__((ext_vector_type(8)))  _Float16 v8h;
typedef __attribute__((ext_vector_type(8)))  float    v8f;

__device__ __forceinline__ float silu_f(float x) {
    return x / (1.0f + __expf(-x));
}

__device__ __forceinline__ v16h frag_a(const _Float16* p) {
    // A 16x32 f16 fragment: per lane two contiguous 16B chunks
    v8h lo = *(const v8h*)(p);
    v8h hi = *(const v8h*)(p + 16);
    return __builtin_shufflevector(lo, hi, 0,1,2,3,4,5,6,7,8,9,10,11,12,13,14,15);
}
__device__ __forceinline__ v16h frag_b(const _Float16* p) {
    // B 32x16 f16 fragment from row-major (N,K) weights: per lane two contiguous 16B chunks
    v8h lo = *(const v8h*)(p);
    v8h hi = *(const v8h*)(p + 16);
    return __builtin_shufflevector(lo, hi, 0,1,2,3,4,5,6,7,8,9,10,11,12,13,14,15);
}

// ---------------------------------------------------------------------------
// K1: txt[b,d] = text_tokens[b,0,:] . W_txt[d,:] + b_txt[d]
// ---------------------------------------------------------------------------
__global__ void txt_kernel(const float* __restrict__ text_tokens,
                           const float* __restrict__ W_txt,
                           const float* __restrict__ b_txt,
                           float* __restrict__ txt) {
    int idx = blockIdx.x * blockDim.x + threadIdx.x;
    if (idx >= B_ * D_MODEL_) return;
    int b = idx / D_MODEL_;
    int d = idx - b * D_MODEL_;
    const float* t = text_tokens + (size_t)b * 128 * D_MODEL_;  // [:,0,:]
    const float* w = W_txt + (size_t)d * D_MODEL_;
    float acc = b_txt[d];
    for (int k = 0; k < D_MODEL_; ++k) acc += t[k] * w[k];
    txt[idx] = acc;
}

// ---------------------------------------------------------------------------
// K2a: tokH[m,k] = f16(image[m,k] + txt[b,k])
// ---------------------------------------------------------------------------
__global__ void pack_tokens_kernel(const float* __restrict__ image,
                                   const float* __restrict__ txt,
                                   _Float16* __restrict__ tokH) {
    int k = blockIdx.x * blockDim.x + threadIdx.x;
    int m = blockIdx.y;
    if (k >= D_MODEL_) return;
    int b = m >> 10;
    float v = image[(size_t)m * D_MODEL_ + k] + txt[(size_t)b * D_MODEL_ + k];
    tokH[(size_t)m * D_MODEL_ + k] = (_Float16)v;
}

// K2b: generic f32 -> f16 conversion
__global__ void cvt_f16_kernel(const float* __restrict__ src,
                               _Float16* __restrict__ dst, int n) {
    int i = blockIdx.x * blockDim.x + threadIdx.x;
    if (i < n) dst[i] = (_Float16)src[i];
}

// ---------------------------------------------------------------------------
// K3: xz[m,n] = sum_k tokH[m,k] * in_projH[n,k]    M=4096 N=3072 K=768
// Register-blocked: one wave -> 32(M) x 64(N) macro-tile via 2x4 accumulators.
// 4 waves/block -> 32 x 256 per block.
// ---------------------------------------------------------------------------
__global__ __launch_bounds__(128) void gemm_xz_kernel(
        const _Float16* __restrict__ A,   // (4096,768)
        const _Float16* __restrict__ W,   // (3072,768) row-major (N,K)
        float* __restrict__ C) {          // (4096,3072)
    const int lane  = threadIdx.x & 31;
    const int wave  = threadIdx.x >> 5;
    const int mbase = blockIdx.x * 32;
    const int nbase = (blockIdx.y * 4 + wave) * 64;
    const int rc    = lane & 15;     // A-row / B-col / C-col within tile
    const int hi    = lane >> 4;     // lane half
    const int koffA = hi * 8;
    const int koffB = hi * 8;
    const _Float16* pa0 = A + (size_t)(mbase + rc) * D_MODEL_ + koffA;
    const _Float16* pa1 = pa0 + (size_t)16 * D_MODEL_;
    const _Float16* pb0 = W + (size_t)(nbase + rc) * D_MODEL_ + koffB;

    v8f acc[2][4] = {};
    for (int k = 0; k < D_MODEL_; k += 32) {
        v16h av0 = frag_a(pa0 + k);
        v16h av1 = frag_a(pa1 + k);
        v16h bv[4];
        #pragma unroll
        for (int t = 0; t < 4; ++t)
            bv[t] = frag_b(pb0 + (size_t)t * 16 * D_MODEL_ + k);
        #pragma unroll
        for (int t = 0; t < 4; ++t) {
            acc[0][t] = __builtin_amdgcn_wmma_f32_16x16x32_f16(
                false, av0, false, bv[t], (short)0, acc[0][t], false, false);
            acc[1][t] = __builtin_amdgcn_wmma_f32_16x16x32_f16(
                false, av1, false, bv[t], (short)0, acc[1][t], false, false);
        }
    }
    #pragma unroll
    for (int mt = 0; mt < 2; ++mt) {
        float* pc = C + (size_t)(mbase + mt * 16 + hi * 8) * XZ_N + nbase + rc;
        #pragma unroll
        for (int t = 0; t < 4; ++t)
            #pragma unroll
            for (int j = 0; j < 8; ++j)
                pc[(size_t)j * XZ_N + t * 16] = acc[mt][t][j];
    }
}

// ---------------------------------------------------------------------------
// K4: depthwise causal conv(4) + bias + SiLU on x half of xz.
// Writes xc (f32) and xcH (f16).
// ---------------------------------------------------------------------------
__global__ void conv_silu_kernel(const float* __restrict__ xz,
                                 const float* __restrict__ conv_w,  // (1536,1,4)
                                 const float* __restrict__ conv_b,
                                 float* __restrict__ xc,
                                 _Float16* __restrict__ xcH) {
    int d = blockIdx.x * blockDim.x + threadIdx.x;
    int m = blockIdx.y;
    if (d >= D_INNER_) return;
    int b = m >> 10, l = m & 1023;
    float acc = conv_b[d];
    #pragma unroll
    for (int j = 0; j < D_CONV_; ++j) {
        int ll = l - (D_CONV_ - 1) + j;
        if (ll >= 0)
            acc += xz[(size_t)(b * L_ + ll) * XZ_N + d] * conv_w[d * D_CONV_ + j];
    }
    float v = silu_f(acc);
    xc[(size_t)m * D_INNER_ + d]  = v;
    xcH[(size_t)m * D_INNER_ + d] = (_Float16)v;
}

// ---------------------------------------------------------------------------
// K5: x_dbl[m,r] = sum_d xcH[m,d] * x_projH[r,d]   M=4096 N=80 K=1536
// 5 waves/block, one 16-col tile each.
// ---------------------------------------------------------------------------
__global__ __launch_bounds__(160) void gemm_xdbl_kernel(
        const _Float16* __restrict__ A,   // (4096,1536)
        const _Float16* __restrict__ W,   // (80,1536)
        float* __restrict__ C) {          // (4096,80)
    const int lane  = threadIdx.x & 31;
    const int wave  = threadIdx.x >> 5;
    const int mbase = blockIdx.x * 16;
    const int nbase = wave * 16;
    const int rc    = lane & 15;
    const int hi    = lane >> 4;
    const _Float16* pa = A + (size_t)(mbase + rc) * D_INNER_ + hi * 8;
    const _Float16* pb = W + (size_t)(nbase + rc) * D_INNER_ + hi * 8;
    v8f acc = {};
    for (int k = 0; k < D_INNER_; k += 32) {
        v16h av = frag_a(pa + k);
        v16h bv = frag_b(pb + k);
        acc = __builtin_amdgcn_wmma_f32_16x16x32_f16(false, av, false, bv,
                                                     (short)0, acc, false, false);
    }
    float* pc = C + (size_t)(mbase + hi * 8) * XDBL_N + nbase + rc;
    #pragma unroll
    for (int j = 0; j < 8; ++j) pc[(size_t)j * XDBL_N] = acc[j];
}

// ---------------------------------------------------------------------------
// K6: dt[m,d] = softplus( x_dbl[m,0:48] . dt_proj_w[d,:] + dt_proj_b[d] )
// ---------------------------------------------------------------------------
__global__ void dt_kernel(const float* __restrict__ xdbl,
                          const float* __restrict__ dt_proj_w,  // (1536,48)
                          const float* __restrict__ dt_proj_b,
                          float* __restrict__ dt) {
    int d = blockIdx.x * blockDim.x + threadIdx.x;
    int m = blockIdx.y;
    if (d >= D_INNER_) return;
    const float* xr = xdbl + (size_t)m * XDBL_N;
    const float* wr = dt_proj_w + (size_t)d * DT_RANK_;
    float acc = dt_proj_b[d];
    #pragma unroll 8
    for (int r = 0; r < DT_RANK_; ++r) acc += xr[r] * wr[r];
    float sp = (acc > 20.0f) ? acc : log1pf(__expf(acc));
    dt[(size_t)m * D_INNER_ + d] = sp;
}

// ---------------------------------------------------------------------------
// K7: selective scan, fused with +u*D, *silu(z) and the mean over L.
// Thread = (b, d, s). 16-lane shfl reduction over s. Writes ybar (4,1536).
// ---------------------------------------------------------------------------
__global__ __launch_bounds__(256) void scan_kernel(
        const float* __restrict__ dt,    // (4096,1536)
        const float* __restrict__ xc,    // (4096,1536) post conv+silu (u)
        const float* __restrict__ xdbl,  // (4096,80): [48:64]=B, [64:80]=C
        const float* __restrict__ xz,    // (4096,3072): z at col 1536+d
        const float* __restrict__ A_log, // (1536,16)
        const float* __restrict__ Dp,    // (1536,)
        float* __restrict__ ybar) {      // (4,1536)
    const int blocksPerBatch = (D_INNER_ * D_STATE_) / 256;  // 96
    const int b = blockIdx.x / blocksPerBatch;
    const int w = blockIdx.x % blocksPerBatch;
    const int s = threadIdx.x & 15;
    const int d = w * 16 + (threadIdx.x >> 4);
    const float Aval = -__expf(A_log[(size_t)d * D_STATE_ + s]);
    const float Dd   = Dp[d];
    float h = 0.0f;
    float yacc = 0.0f;
    for (int l = 0; l < L_; ++l) {
        const size_t m = (size_t)b * L_ + l;
        float dtv = dt[m * D_INNER_ + d];   // broadcast across the 16 s-lanes
        float u   = xc[m * D_INNER_ + d];
        float Bs  = xdbl[m * XDBL_N + DT_RANK_ + s];
        float Cs  = xdbl[m * XDBL_N + DT_RANK_ + D_STATE_ + s];
        h = __expf(dtv * Aval) * h + (dtv * u) * Bs;
        float p = h * Cs;
        // sum over the 16 states (lanes [0..15] / [16..31] of the wave)
        p += __shfl_xor(p, 1, 32);
        p += __shfl_xor(p, 2, 32);
        p += __shfl_xor(p, 4, 32);
        p += __shfl_xor(p, 8, 32);
        if (s == 0) {
            float y = p + u * Dd;
            float z = xz[m * XZ_N + D_INNER_ + d];
            y *= silu_f(z);
            yacc += y;
        }
    }
    if (s == 0) ybar[(size_t)b * D_INNER_ + d] = yacc * (1.0f / (float)L_);
}

// ---------------------------------------------------------------------------
// K8: out[b,n] = ybar[b,:] . out_proj_w[n,:]   (mean commuted before GEMM)
// ---------------------------------------------------------------------------
__global__ void out_kernel(const float* __restrict__ ybar,
                           const float* __restrict__ out_proj_w, // (768,1536)
                           float* __restrict__ out) {
    int idx = blockIdx.x * blockDim.x + threadIdx.x;
    if (idx >= B_ * D_MODEL_) return;
    int b = idx / D_MODEL_;
    int n = idx - b * D_MODEL_;
    const float* y = ybar + (size_t)b * D_INNER_;
    const float* w = out_proj_w + (size_t)n * D_INNER_;
    float acc = 0.0f;
    for (int d = 0; d < D_INNER_; ++d) acc += y[d] * w[d];
    out[idx] = acc;
}

// ---------------------------------------------------------------------------
// Workspace layout (bytes)
// ---------------------------------------------------------------------------
#define OFF_TXT   ((size_t)0)                                   // 4*768*4
#define OFF_TOKH  (OFF_TXT  + (size_t)B_ * D_MODEL_ * 4)        // 4096*768*2
#define OFF_WINH  (OFF_TOKH + (size_t)M_TOT * D_MODEL_ * 2)     // 3072*768*2
#define OFF_XPH   (OFF_WINH + (size_t)XZ_N * D_MODEL_ * 2)      // 80*1536*2
#define OFF_XZ    (OFF_XPH  + (size_t)XDBL_N * D_INNER_ * 2)    // 4096*3072*4
#define OFF_XC    (OFF_XZ   + (size_t)M_TOT * XZ_N * 4)         // 4096*1536*4
#define OFF_XCH   (OFF_XC   + (size_t)M_TOT * D_INNER_ * 4)     // 4096*1536*2
#define OFF_XDBL  (OFF_XCH  + (size_t)M_TOT * D_INNER_ * 2)     // 4096*80*4
#define OFF_DT    (OFF_XDBL + (size_t)M_TOT * XDBL_N * 4)       // 4096*1536*4
#define OFF_YBAR  (OFF_DT   + (size_t)M_TOT * D_INNER_ * 4)     // 4*1536*4

extern "C" void kernel_launch(void* const* d_in, const int* in_sizes, int n_in,
                              void* d_out, int out_size, void* d_ws, size_t ws_size,
                              hipStream_t stream) {
    const float* image_tokens = (const float*)d_in[0];
    const float* text_tokens  = (const float*)d_in[1];
    const float* W_txt        = (const float*)d_in[2];
    const float* b_txt        = (const float*)d_in[3];
    const float* in_proj_w    = (const float*)d_in[4];
    const float* conv_w       = (const float*)d_in[5];
    const float* conv_b       = (const float*)d_in[6];
    const float* x_proj_w     = (const float*)d_in[7];
    const float* dt_proj_w    = (const float*)d_in[8];
    const float* dt_proj_b    = (const float*)d_in[9];
    const float* A_log        = (const float*)d_in[10];
    const float* Dp           = (const float*)d_in[11];
    const float* out_proj_w   = (const float*)d_in[12];
    float* out = (float*)d_out;

    char* ws = (char*)d_ws;
    float*    txt   = (float*)   (ws + OFF_TXT);
    _Float16* tokH  = (_Float16*)(ws + OFF_TOKH);
    _Float16* wInH  = (_Float16*)(ws + OFF_WINH);
    _Float16* xpH   = (_Float16*)(ws + OFF_XPH);
    float*    xz    = (float*)   (ws + OFF_XZ);
    float*    xc    = (float*)   (ws + OFF_XC);
    _Float16* xcH   = (_Float16*)(ws + OFF_XCH);
    float*    xdbl  = (float*)   (ws + OFF_XDBL);
    float*    dt    = (float*)   (ws + OFF_DT);
    float*    ybar  = (float*)   (ws + OFF_YBAR);

    // K1: text projection
    txt_kernel<<<(B_ * D_MODEL_ + 255) / 256, 256, 0, stream>>>(
        text_tokens, W_txt, b_txt, txt);

    // K2: pack tokens + weight conversions to f16
    pack_tokens_kernel<<<dim3(D_MODEL_ / 256, M_TOT), 256, 0, stream>>>(
        image_tokens, txt, tokH);
    cvt_f16_kernel<<<(XZ_N * D_MODEL_ + 255) / 256, 256, 0, stream>>>(
        in_proj_w, wInH, XZ_N * D_MODEL_);
    cvt_f16_kernel<<<(XDBL_N * D_INNER_ + 255) / 256, 256, 0, stream>>>(
        x_proj_w, xpH, XDBL_N * D_INNER_);

    // K3: big WMMA GEMM  xz = tokens @ in_proj_w.T  (32x256 per block)
    gemm_xz_kernel<<<dim3(M_TOT / 32, XZ_N / 256), 128, 0, stream>>>(tokH, wInH, xz);

    // K4: depthwise conv + SiLU
    conv_silu_kernel<<<dim3(D_INNER_ / 256, M_TOT), 256, 0, stream>>>(
        xz, conv_w, conv_b, xc, xcH);

    // K5: WMMA GEMM  x_dbl = x @ x_proj_w.T
    gemm_xdbl_kernel<<<dim3(M_TOT / 16), 160, 0, stream>>>(xcH, xpH, xdbl);

    // K6: dt projection + softplus
    dt_kernel<<<dim3(D_INNER_ / 256, M_TOT), 256, 0, stream>>>(
        xdbl, dt_proj_w, dt_proj_b, dt);

    // K7: selective scan fused with gating + L-mean
    scan_kernel<<<(B_ * D_INNER_ * D_STATE_) / 256, 256, 0, stream>>>(
        dt, xc, xdbl, xz, A_log, Dp, ybar);

    // K8: final projection of the mean
    out_kernel<<<(B_ * D_MODEL_ + 255) / 256, 256, 0, stream>>>(
        ybar, out_proj_w, out);
}
